// NnBoard768_22720376996044
// MI455X (gfx1250) — matrix operand order repaired and checked by reading, not compiled, not run on hardware
//
#include <hip/hip_runtime.h>
#include <hip/hip_bf16.h>

// ---- problem constants (match reference) ----
#define BATCH  16384
#define MAXACT 32
#define FEAT   768
#define HIDDEN 1024
#define PAD    776          // 768 + 8 halfs: row padding to break LDS bank conflicts
#define KT     (FEAT / 32)  // 24 K-tiles of 32
#define NT     (HIDDEN / 16)// 64 N-tiles of 16
#define NTW    8            // N-tiles per wave (8 waves * 8 = 64)
#define TS     2            // sample tiles (of 16) per workgroup

typedef __attribute__((ext_vector_type(16))) _Float16 v16h;
typedef __attribute__((ext_vector_type(8)))  _Float16 v8h;
typedef __attribute__((ext_vector_type(8)))  float    v8f;

// ------------------------------------------------------------------
// Kernel 1: ft_w (f32, [H][F], torch layout) -> f16 WMMA-B layout.
// B-fragment for tile (nt,kt): lane n<16 : halfs j=0..15 = B[K=kt*32+j][N=nt*16+n]
//                              lane n>=16: halfs j=0..15 = B[K=kt*32+16+j][N=nt*16+(n-16)]
// with B[f][h] = ft_w[h][f]. Stored so each lane's v16h is 32 contiguous bytes.
// ------------------------------------------------------------------
__global__ void nn_prep_weights(const float* __restrict__ ft_w,
                                _Float16* __restrict__ wB) {
    int t = blockIdx.x * blockDim.x + threadIdx.x;
    if (t >= HIDDEN * FEAT) return;
    int j    = t & 15;
    int lane = (t >> 4) & 31;
    int ktnt = t >> 9;
    int kt   = ktnt % KT;
    int nt   = ktnt / KT;
    int h = nt * 16 + (lane & 15);
    int f = kt * 32 + ((lane >> 4) ? 16 : 0) + j;
    wB[t] = (_Float16)ft_w[h * FEAT + f];
}

// ------------------------------------------------------------------
// Kernel 2: per (pair of 16-sample tiles, side): densify one-hot features
// into LDS f16 A-tiles, sweep kt outermost so each A fragment feeds all 8
// resident nt accumulators, and each B fragment feeds both sample tiles:
// 16 v_wmma per kt step vs 4 ds_load_b128 + 16 global_load_b128.
// ------------------------------------------------------------------
__global__ __launch_bounds__(256)
void nn_ft_wmma(const int* __restrict__ stm, const int* __restrict__ nstm,
                const _Float16* __restrict__ wB, const float* __restrict__ ft_b,
                const float* __restrict__ out_w, float* __restrict__ partial) {
    __shared__ __align__(16) _Float16 tileA[TS][16 * PAD];
    __shared__ float wave_part[8][TS][16];

    const int side = blockIdx.x & 1;
    const int pair = blockIdx.x >> 1;           // covers samples pair*32 .. pair*32+31
    const int tid  = threadIdx.x;
    const int* __restrict__ ind = side ? nstm : stm;

    // zero the dense tiles (dword stores)
    for (int i = tid; i < TS * 16 * PAD / 2; i += 256)
        ((unsigned*)tileA)[i] = 0u;
    __syncthreads();

    // scatter ones: TS*16 samples x 32 active indices (-1 = padding)
    for (int i = tid; i < TS * 16 * MAXACT; i += 256) {
        int s = i >> 5, k = i & 31;             // s in [0, 32): local sample
        int idx = ind[(pair * (TS * 16) + s) * MAXACT + k];
        if (idx >= 0 && idx < FEAT)
            tileA[s >> 4][(s & 15) * PAD + idx] = (_Float16)1.0f;
    }
    __syncthreads();

    const int lane   = tid & 31;
    const int wave   = tid >> 5;
    const int l16    = lane & 15;
    const int hiHalf = lane >> 4;   // 0: K {0..15}; 1: K {16..31}

    // accumulators: c[i][t] for nt = i*8 + wave, sample tile t. D layout:
    // lanes<16: (M=r, N=l16); lanes>=16: (M=r+8, N=l16). Init with ft_b[h].
    v8f c[NTW][TS];
#pragma unroll
    for (int i = 0; i < NTW; ++i) {
        const float bias = ft_b[(i * 8 + wave) * 16 + l16];
#pragma unroll
        for (int t = 0; t < TS; ++t)
#pragma unroll
            for (int r = 0; r < 8; ++r) c[i][t][r] = bias;
    }

    // kt outermost: A fragments (per sample tile) reused by all 8 nt tiles.
    for (int kt = 0; kt < KT; ++kt) {
        const int colbase = kt * 32 + hiHalf * 16;
        v16h a[TS];
#pragma unroll
        for (int t = 0; t < TS; ++t) {
            const _Float16* rowA = &tileA[t][l16 * PAD];
            v8h alo = *(const v8h*)(rowA + colbase);        // K base..base+7
            v8h ahi = *(const v8h*)(rowA + colbase + 8);    // K base+8..base+15
            a[t] = __builtin_shufflevector(alo, ahi,
                      0,1,2,3,4,5,6,7,8,9,10,11,12,13,14,15);
        }
#pragma unroll
        for (int i = 0; i < NTW; ++i) {
            const int nt = i * 8 + wave;
            v16h b = *(const v16h*)(wB + (((nt * KT + kt) * 32 + lane) << 4));
#pragma unroll
            for (int t = 0; t < TS; ++t)
                c[i][t] = __builtin_amdgcn_wmma_f32_16x16x32_f16(
                              /*neg_a=*/false, a[t], /*neg_b=*/false, b,
                              /*c_mod=*/(short)0, c[i][t],
                              /*reuse_a=*/false, /*reuse_b=*/false);
        }
    }

    // epilogue: clipped ReLU + output-layer contraction into per-sample acc
    float acc[TS][8];
#pragma unroll
    for (int t = 0; t < TS; ++t)
#pragma unroll
        for (int r = 0; r < 8; ++r) acc[t][r] = 0.f;

#pragma unroll
    for (int i = 0; i < NTW; ++i) {
        const float w = out_w[side * HIDDEN + (i * 8 + wave) * 16 + l16];
#pragma unroll
        for (int t = 0; t < TS; ++t)
#pragma unroll
            for (int r = 0; r < 8; ++r)
                acc[t][r] += fminf(fmaxf(c[i][t][r], 0.f), 1.f) * w;
    }

    // deterministic reduction over the 16 N-lanes of each half-wave
#pragma unroll
    for (int m = 1; m < 16; m <<= 1)
#pragma unroll
        for (int t = 0; t < TS; ++t)
#pragma unroll
            for (int r = 0; r < 8; ++r)
                acc[t][r] += __shfl_xor(acc[t][r], m, 32);

    if (lane == 0)
#pragma unroll
        for (int t = 0; t < TS; ++t)
#pragma unroll
            for (int r = 0; r < 8; ++r) wave_part[wave][t][r] = acc[t][r];      // M=r
    if (lane == 16)
#pragma unroll
        for (int t = 0; t < TS; ++t)
#pragma unroll
            for (int r = 0; r < 8; ++r) wave_part[wave][t][8 + r] = acc[t][r];  // M=r+8
    __syncthreads();

    if (tid < TS * 16) {
        const int t = tid >> 4, s = tid & 15;
        float p = 0.f;
#pragma unroll
        for (int w8 = 0; w8 < 8; ++w8) p += wave_part[w8][t][s];
        partial[side * BATCH + pair * (TS * 16) + t * 16 + s] = p;  // exclusive slot
    }
}

// ------------------------------------------------------------------
// Kernel 3: out = sigmoid(partial_stm + partial_nstm + out_b)
// ------------------------------------------------------------------
__global__ void nn_finalize(const float* __restrict__ partial,
                            const float* __restrict__ out_b,
                            float* __restrict__ out) {
    int b = blockIdx.x * blockDim.x + threadIdx.x;
    if (b >= BATCH) return;
    float x = partial[b] + partial[BATCH + b] + out_b[0];
    out[b] = 1.0f / (1.0f + __expf(-x));
}

extern "C" void kernel_launch(void* const* d_in, const int* in_sizes, int n_in,
                              void* d_out, int out_size, void* d_ws, size_t ws_size,
                              hipStream_t stream) {
    const int*   stm   = (const int*)d_in[0];
    const int*   nstm  = (const int*)d_in[1];
    const float* ft_w  = (const float*)d_in[2];
    const float* ft_b  = (const float*)d_in[3];
    const float* out_w = (const float*)d_in[4];
    const float* out_b = (const float*)d_in[5];

    _Float16* wB      = (_Float16*)d_ws;                                   // 1.5 MB
    float*    partial = (float*)((char*)d_ws + (size_t)HIDDEN * FEAT * 2); // 128 KB

    nn_prep_weights<<<(HIDDEN * FEAT + 255) / 256, 256, 0, stream>>>(ft_w, wB);
    nn_ft_wmma<<<(BATCH / (16 * TS)) * 2, 256, 0, stream>>>(stm, nstm, wB, ft_b,
                                                            out_w, partial);
    nn_finalize<<<(BATCH + 255) / 256, 256, 0, stream>>>(partial, out_b, (float*)d_out);
}
